// MultiScaleDeformableAttention_82884278879142
// MI455X (gfx1250) — matrix operand order repaired and checked by reading, not compiled, not run on hardware
//
#include <hip/hip_runtime.h>

// Multi-Scale Deformable Attention for MI455X (gfx1250, wave32, WMMA).
// bs=1, nq=21760, EMBED=256, 8 heads x 32 dim, 4 levels x 4 points.
// All GEMMs run on v_wmma_f32_16x16x32_bf16 with pre-packed bf16 operands:
// hot loops are pure b128-load + wmma (conversions hoisted to a prep pass
// using single v_perm_b32 packs).

#define NQ     21760
#define EMBEDD 256
#define NHEAD  8
#define NLVL   4
#define NPTS   4
#define HDIM   32

typedef __attribute__((ext_vector_type(16))) __bf16 v16bf;
typedef __attribute__((ext_vector_type(8)))  float  v8f;

// ---- bf16 pair packing: one v_add per float + one v_perm_b32 per pair ------
__device__ __forceinline__ unsigned pk2bf(float a, float b) {
  const unsigned ua = __float_as_uint(a) + 0x8000u;   // round-half-up to bf16
  const unsigned ub = __float_as_uint(b) + 0x8000u;
  // result: [15:0] = ua[31:16], [31:16] = ub[31:16]
  return __builtin_amdgcn_perm(ub, ua, 0x07060302u);
}

union FragBF { unsigned u[8]; v16bf v; };

// ---- WMMA fragment loaders from packed-bf16 (1 dword per K-pair) ----------
// A (16x32, MxK): lane 0-15 M=lane K groups {16h..16h+7},{16h+8..16h+15}.
// Packed row stride = ldp dwords. Two uint4 loads, zero VALU.
__device__ __forceinline__ v16bf load_a_pk(const unsigned* Apk, int ldp,
                                           int m0, int k0, int lane) {
  const int half = lane >> 4;
  const unsigned* row = Apk + (size_t)(m0 + (lane & 15)) * ldp;
  const uint4 g1 = *(const uint4*)(row + ((k0 + 16 * half) >> 1));
  const uint4 g2 = *(const uint4*)(row + ((k0 + 16 * half + 8) >> 1));
  FragBF f;
  f.u[0] = g1.x; f.u[1] = g1.y; f.u[2] = g1.z; f.u[3] = g1.w;
  f.u[4] = g2.x; f.u[5] = g2.y; f.u[6] = g2.z; f.u[7] = g2.w;
  return f.v;
}

// B (32x16, KxN) with B[k][n] = W[n][k]: VGPR j holds K=16h+2j,16h+2j+1 at
// N=lane%16 -> 8 consecutive packed dwords of row n. Two uint4 loads.
__device__ __forceinline__ v16bf load_b_pk(const unsigned* Wpk, int ldp,
                                           int n0, int k0, int lane) {
  const int half = lane >> 4;
  const unsigned* row = Wpk + (size_t)(n0 + (lane & 15)) * ldp
                            + ((k0 + 16 * half) >> 1);
  const uint4 g1 = *(const uint4*)(row);
  const uint4 g2 = *(const uint4*)(row + 4);
  FragBF f;
  f.u[0] = g1.x; f.u[1] = g1.y; f.u[2] = g1.z; f.u[3] = g1.w;
  f.u[4] = g2.x; f.u[5] = g2.y; f.u[6] = g2.z; f.u[7] = g2.w;
  return f.v;
}

// ---- Prep: pack f32 pairs to bf16x2 dwords ---------------------------------
__global__ void pack_bf16_pairs(const float* __restrict__ src,
                                unsigned* __restrict__ dst, int npairs) {
  const int i = blockIdx.x * blockDim.x + threadIdx.x;
  if (i < npairs) {
    const float2 f = ((const float2*)src)[i];
    dst[i] = pk2bf(f.x, f.y);
  }
}

// ---- Kernel 1: v = value @ W_value^T + b_value (bf16 WMMA, f32 out) --------
// grid = (NQ/16, 4), block = 128 (4 waves); each wave owns one 16x16 tile.
__global__ void msda_value_proj(const unsigned* __restrict__ value_pk,
                                const unsigned* __restrict__ Wv_pk,
                                const float* __restrict__ bv,
                                float* __restrict__ vout) {
  const int lane = threadIdx.x & 31;
  const int wave = threadIdx.x >> 5;
  const int m0 = blockIdx.x * 16;
  const int n0 = (blockIdx.y * 4 + wave) * 16;
  v8f acc = {};
#pragma unroll
  for (int k0 = 0; k0 < EMBEDD; k0 += 32) {
    if (k0 + 32 < EMBEDD)  // global_prefetch_b8 of next A slab
      __builtin_prefetch(value_pk + (size_t)(m0 + (lane & 15)) * (EMBEDD / 2)
                         + ((k0 + 32) >> 1), 0, 1);
    v16bf a = load_a_pk(value_pk, EMBEDD / 2, m0, k0, lane);
    v16bf b = load_b_pk(Wv_pk, EMBEDD / 2, n0, k0, lane);
    acc = __builtin_amdgcn_wmma_f32_16x16x32_bf16(false, a, false, b,
                                                  (short)0, acc, false, false);
  }
  const int n = n0 + (lane & 15);
  const float bias = bv[n];
#pragma unroll
  for (int r = 0; r < 8; ++r) {
    const int m = m0 + r + 8 * (lane >> 4);
    vout[(size_t)m * EMBEDD + n] = acc[r] + bias;
  }
}

// ---- Kernel 2: fused q-add, offset/attn GEMMs, softmax, bilinear sampling --
// grid = (NQ/16), block = 256 (8 waves). LDS ~32.5 KB.
__global__ void msda_sample(const float* __restrict__ query,
                            const float* __restrict__ query_pos,
                            const float* __restrict__ refpts,
                            const unsigned* __restrict__ Woff_pk,
                            const float* __restrict__ boff,
                            const unsigned* __restrict__ Wattn_pk,
                            const float* __restrict__ battn,
                            const float* __restrict__ v,
                            unsigned* __restrict__ sampled_pk) {
  __shared__ unsigned q_pk[16][EMBEDD / 2];           // q tile as packed bf16
  __shared__ float    off_s[16][EMBEDD];              // offsets (f32)
  __shared__ float    at_s[16][NHEAD * NLVL * NPTS];  // attn weights (f32)
  __shared__ float    ref_s[16][NLVL * 2];            // reference points

  const int tid  = threadIdx.x;
  const int lane = tid & 31;
  const int wave = tid >> 5;
  const int m0   = blockIdx.x * 16;

  // Stage 1: q = query + query_pos, packed to bf16 pairs in LDS (v_perm_b32).
  {
    const float2* qg = (const float2*)(query     + (size_t)m0 * EMBEDD);
    const float2* qp = (const float2*)(query_pos + (size_t)m0 * EMBEDD);
    for (int i = tid; i < 16 * (EMBEDD / 2); i += blockDim.x) {
      const float2 a = qg[i], b = qp[i];
      q_pk[i >> 7][i & 127] = pk2bf(a.x + b.x, a.y + b.y);
    }
    if (tid < 16 * NLVL * 2)
      (&ref_s[0][0])[tid] = refpts[(size_t)m0 * NLVL * 2 + tid];
  }
  __syncthreads();

  // Stage 2: 24 WMMA n-tiles (16 for offsets N=256, 8 for attn N=128).
  const unsigned* qpk_flat = &q_pk[0][0];
  for (int t = wave; t < 24; t += 8) {   // uniform per wave -> EXEC all 1s
    const bool isOff      = (t < 16);
    const unsigned* Wpk   = isOff ? Woff_pk : Wattn_pk;
    const float* bias     = isOff ? boff : battn;
    const int n0          = isOff ? t * 16 : (t - 16) * 16;
    v8f acc = {};
#pragma unroll
    for (int k0 = 0; k0 < EMBEDD; k0 += 32) {
      v16bf a = load_a_pk(qpk_flat, EMBEDD / 2, 0, k0, lane);
      v16bf b = load_b_pk(Wpk, EMBEDD / 2, n0, k0, lane);
      acc = __builtin_amdgcn_wmma_f32_16x16x32_bf16(false, a, false, b,
                                                    (short)0, acc, false, false);
    }
    const int nl = n0 + (lane & 15);
    const float bb = bias[nl];
#pragma unroll
    for (int r = 0; r < 8; ++r) {
      const int mr = r + 8 * (lane >> 4);
      if (isOff) off_s[mr][nl] = acc[r] + bb;
      else       at_s[mr][nl]  = acc[r] + bb;
    }
  }
  __syncthreads();

  // Stage 3: softmax over 16 (lvl,pt) per (query, head).
  if (tid < 16 * NHEAD) {
    const int qr = tid >> 3, h = tid & 7;
    float* a = &at_s[qr][h * 16];
    float mx = a[0];
#pragma unroll
    for (int i = 1; i < 16; ++i) mx = fmaxf(mx, a[i]);
    float s = 0.f;
#pragma unroll
    for (int i = 0; i < 16; ++i) { const float e = __expf(a[i] - mx); a[i] = e; s += e; }
    const float inv = 1.0f / s;
#pragma unroll
    for (int i = 0; i < 16; ++i) a[i] *= inv;
  }
  __syncthreads();

  // Stage 4: bilinear gather. One wave per (query, head); lane = channel.
  // Locations are lane-uniform -> scalar branches; gathers are 128B coalesced.
  const int Hs[4] = {128, 64, 32, 16};
  const int Ls[4] = {0, 16384, 20480, 21504};
  for (int t = wave; t < 16 * NHEAD; t += 8) {
    const int qr = t >> 3, h = t & 7;
    float acc = 0.f;
    const size_t col = (size_t)h * HDIM + lane;
#pragma unroll
    for (int lvl = 0; lvl < NLVL; ++lvl) {
      const int Hl = Hs[lvl], Wl = Hs[lvl], base = Ls[lvl];
      const float Hf = (float)Hl, Wf = (float)Wl;
      const float rx = ref_s[qr][lvl * 2 + 0];
      const float ry = ref_s[qr][lvl * 2 + 1];
#pragma unroll
      for (int p = 0; p < NPTS; ++p) {
        const int oi = ((h * NLVL + lvl) * NPTS + p) * 2;
        const float aw = at_s[qr][(h * NLVL + lvl) * NPTS + p];
        // grid = 2*loc-1; gx = (grid+1)*W/2-0.5 simplifies to loc*W-0.5
        const float gx = (rx + off_s[qr][oi]     / Wf) * Wf - 0.5f;
        const float gy = (ry + off_s[qr][oi + 1] / Hf) * Hf - 0.5f;
        const float x0f = floorf(gx), y0f = floorf(gy);
        const int   x0 = (int)x0f,   y0 = (int)y0f;
        const float wx = gx - x0f,   wy = gy - y0f;
        const float w00 = (1.f - wx) * (1.f - wy) * aw;
        const float w10 = wx * (1.f - wy) * aw;
        const float w01 = (1.f - wx) * wy * aw;
        const float w11 = wx * wy * aw;
        if ((unsigned)x0 < (unsigned)Wl && (unsigned)y0 < (unsigned)Hl)
          acc += w00 * v[(size_t)(base + y0 * Wl + x0) * EMBEDD + col];
        if ((unsigned)(x0 + 1) < (unsigned)Wl && (unsigned)y0 < (unsigned)Hl)
          acc += w10 * v[(size_t)(base + y0 * Wl + x0 + 1) * EMBEDD + col];
        if ((unsigned)x0 < (unsigned)Wl && (unsigned)(y0 + 1) < (unsigned)Hl)
          acc += w01 * v[(size_t)(base + (y0 + 1) * Wl + x0) * EMBEDD + col];
        if ((unsigned)(x0 + 1) < (unsigned)Wl && (unsigned)(y0 + 1) < (unsigned)Hl)
          acc += w11 * v[(size_t)(base + (y0 + 1) * Wl + x0 + 1) * EMBEDD + col];
      }
    }
    // Write sampled directly as packed bf16 (pairs combined across lanes).
    const float accN = __shfl_down(acc, 1, 32);
    if ((lane & 1) == 0)
      sampled_pk[((size_t)(m0 + qr) * EMBEDD + h * HDIM + lane) >> 1] =
          pk2bf(acc, accN);
  }
}

// ---- Kernel 3: out = sampled @ W_out^T + b_out + query (residual) ----------
__global__ void msda_out_proj(const unsigned* __restrict__ sampled_pk,
                              const unsigned* __restrict__ Wo_pk,
                              const float* __restrict__ bo,
                              const float* __restrict__ identity,
                              float* __restrict__ out) {
  const int lane = threadIdx.x & 31;
  const int wave = threadIdx.x >> 5;
  const int m0 = blockIdx.x * 16;
  const int n0 = (blockIdx.y * 4 + wave) * 16;
  v8f acc = {};
#pragma unroll
  for (int k0 = 0; k0 < EMBEDD; k0 += 32) {
    v16bf a = load_a_pk(sampled_pk, EMBEDD / 2, m0, k0, lane);
    v16bf b = load_b_pk(Wo_pk, EMBEDD / 2, n0, k0, lane);
    acc = __builtin_amdgcn_wmma_f32_16x16x32_bf16(false, a, false, b,
                                                  (short)0, acc, false, false);
  }
  const int n = n0 + (lane & 15);
  const float bias = bo[n];
#pragma unroll
  for (int r = 0; r < 8; ++r) {
    const int m = m0 + r + 8 * (lane >> 4);
    out[(size_t)m * EMBEDD + n] = acc[r] + bias + identity[(size_t)m * EMBEDD + n];
  }
}

extern "C" void kernel_launch(void* const* d_in, const int* in_sizes, int n_in,
                              void* d_out, int out_size, void* d_ws, size_t ws_size,
                              hipStream_t stream) {
  (void)in_sizes; (void)n_in; (void)out_size; (void)ws_size;
  const float* query     = (const float*)d_in[0];
  const float* query_pos = (const float*)d_in[1];
  const float* value     = (const float*)d_in[2];
  const float* refpts    = (const float*)d_in[3];
  // d_in[4] = spatial_shapes (int32) — compile-time constants here.
  const float* Wv   = (const float*)d_in[5];
  const float* bv   = (const float*)d_in[6];
  const float* Woff = (const float*)d_in[7];
  const float* boff = (const float*)d_in[8];
  const float* Wat  = (const float*)d_in[9];
  const float* bat  = (const float*)d_in[10];
  const float* Wo   = (const float*)d_in[11];
  const float* bo   = (const float*)d_in[12];
  float* out = (float*)d_out;

  // Workspace layout.
  float*    v_f32    = (float*)d_ws;                                  // 22.3 MB
  unsigned* value_pk = (unsigned*)(v_f32 + (size_t)NQ * EMBEDD);      // 11.1 MB
  unsigned* samp_pk  = value_pk + (size_t)NQ * (EMBEDD / 2);          // 11.1 MB
  unsigned* Wv_pk    = samp_pk  + (size_t)NQ * (EMBEDD / 2);
  unsigned* Woff_pk  = Wv_pk    + EMBEDD * (EMBEDD / 2);
  unsigned* Wat_pk   = Woff_pk  + EMBEDD * (EMBEDD / 2);
  unsigned* Wo_pk    = Wat_pk   + (NHEAD * NLVL * NPTS) * (EMBEDD / 2);

  // Prep: one-time f32 -> packed bf16 conversions (v_perm_b32 packing).
  const int PB = 256;
  const int npV = NQ * (EMBEDD / 2);
  const int npW = EMBEDD * (EMBEDD / 2);
  const int npA = (NHEAD * NLVL * NPTS) * (EMBEDD / 2);
  pack_bf16_pairs<<<(npV + PB - 1) / PB, PB, 0, stream>>>(value, value_pk, npV);
  pack_bf16_pairs<<<(npW + PB - 1) / PB, PB, 0, stream>>>(Wv,   Wv_pk,   npW);
  pack_bf16_pairs<<<(npW + PB - 1) / PB, PB, 0, stream>>>(Woff, Woff_pk, npW);
  pack_bf16_pairs<<<(npA + PB - 1) / PB, PB, 0, stream>>>(Wat,  Wat_pk,  npA);
  pack_bf16_pairs<<<(npW + PB - 1) / PB, PB, 0, stream>>>(Wo,   Wo_pk,   npW);

  const dim3 gGemm(NQ / 16, 4), bGemm(128);            // 4 waves / block
  msda_value_proj<<<gGemm, bGemm, 0, stream>>>(value_pk, Wv_pk, bv, v_f32);
  msda_sample<<<dim3(NQ / 16), dim3(256), 0, stream>>>(query, query_pos, refpts,
                                                       Woff_pk, boff, Wat_pk, bat,
                                                       v_f32, samp_pk);
  msda_out_proj<<<gGemm, bGemm, 0, stream>>>(samp_pk, Wo_pk, bo, query, out);
}
